// Cubify_19069654794323
// MI455X (gfx1250) — compile-verified, matrix operand order checked by reading, not run on hardware
//
#include <hip/hip_runtime.h>

// ---------------- problem constants (fixed by reference: (4,32,32,32)) -----
#define N_    4
#define DIM   32
#define G     33                      // grid verts per axis
#define VG    (G*G*G)                 // 35937 vertices per batch
#define NVOX  (N_*DIM*DIM*DIM)        // 131072
#define NPAIR (NVOX*6)                // 786432 (voxel,face) pairs
#define FCNT  (NPAIR*2)               // 1572864 triangles (F)
#define THR   0.5f

// flat float32 offsets into d_out (reference tuple order)
#define OFF_VC    0
#define OFF_FC    4
#define OFF_VPOS  8
#define OFF_USED  431252              // 8 + 4*VG*3
#define OFF_EDGE  575000              // + 4*VG
#define OFF_FACES 19449368            // + 2*6*FCNT
#define OFF_FMASK 24167960            // + 3*FCNT
#define OFF_EMASK 25740824            // + FCNT

// Per-face triangle vertex deltas in grid-vertex index units
// (dz*1089 + dy*33 + dx), derived from QUAD_OFFS[:,TRI_SEL,:].
__constant__ int TRI_D[6][2][3] = {
  {{0,1,33},        {1,33,34}},
  {{1089,1090,1122},{1090,1122,1123}},
  {{1089,1090,0},   {1090,0,1}},
  {{33,34,1122},    {34,1122,1123}},
  {{1089,0,1122},   {0,1122,33}},
  {{1,1090,34},     {1090,34,1123}}
};
// 4 quad-corner deltas per face (union of both triangles' vertices)
__constant__ int QUAD_D[6][4] = {
  {0,1,33,34},{1089,1090,1122,1123},{1089,1090,0,1},
  {33,34,1122,1123},{1089,0,1122,33},{1,1090,34,1123}
};

typedef __attribute__((ext_vector_type(16))) _Float16 v16h;
typedef __attribute__((ext_vector_type(8)))  float    v8f;

// Wave32 sum-reduction through the matrix core: A = ones(16x32),
// B fragment holds each lane's value in one f16 slot -> every D[m][n] is a
// column sum independent of the K->slot mapping; 4 shuffles finish the wave.
// MUST be called with EXEC all-ones (no divergence at the call site).
__device__ __forceinline__ float wave_sum_wmma(float val) {
  v16h a, b;
#pragma unroll
  for (int i = 0; i < 16; ++i) { a[i] = (_Float16)1.0f; b[i] = (_Float16)0.0f; }
  b[0] = (_Float16)val;
  v8f c = {};
  c = __builtin_amdgcn_wmma_f32_16x16x32_f16(false, a, false, b,
                                             (short)0, c, false, false);
  float s = c[0];                 // lanes 0-15: D[0][n]; lanes 16-31: D[8][n]
  s += __shfl_xor(s, 1, 32);
  s += __shfl_xor(s, 2, 32);
  s += __shfl_xor(s, 4, 32);
  s += __shfl_xor(s, 8, 32);      // now every lane holds the full wave sum
  return s;
}

__device__ __forceinline__ void st2(float* p, float lo, float hi) {
  float2 v; v.x = lo; v.y = hi;
  *reinterpret_cast<float2*>(p) = v;
}

__global__ void k_init(int* __restrict__ ws, int n) {
  int i = blockIdx.x * blockDim.x + threadIdx.x;
  if (i < n) ws[i] = 0;
}

// One thread per (voxel, face). 786432 threads, exact multiple of 256 -> no
// tail, EXEC is all-ones at the WMMA. n is wave-uniform (196608 % 32 == 0).
// Each thread owns triangles j = 2*tid and 2*tid+1, i.e. a pair of
// consecutive elements in every output region -> all stores are float2 (b64).
__global__ void k_faces(const float* __restrict__ probas,
                        float* __restrict__ out,
                        int* __restrict__ wsUsed,
                        int* __restrict__ wsFaceCnt) {
  const int tid = blockIdx.x * blockDim.x + threadIdx.x;   // < NPAIR
  const int f = tid % 6;
  const int v = tid / 6;                                   // linear voxel idx
  const int x = v & 31;
  const int y = (v >> 5) & 31;
  const int z = (v >> 10) & 31;
  const int n = v >> 15;

  const bool occ = probas[v] > THR;

  // neighbor direction: f>>1 selects axis (z,y,x), f&1 selects +/-.
  const int axis = f >> 1;
  const int sgn  = (f & 1) ? 1 : -1;
  const int nz = z + ((axis == 0) ? sgn : 0);
  const int ny = y + ((axis == 1) ? sgn : 0);
  const int nx = x + ((axis == 2) ? sgn : 0);
  bool nocc = false;
  if (((unsigned)nz < 32u) && ((unsigned)ny < 32u) && ((unsigned)nx < 32u))
    nocc = probas[((n * 32 + nz) * 32 + ny) * 32 + nx] > THR;

  const bool exposed = occ && !nocc;
  const float m = exposed ? 1.0f : 0.0f;
  const int baseVid = n * VG + z * 1089 + y * 33 + x;

  // triangle 0 vertices (a*) and triangle 1 vertices (b*) as floats
  const float a0 = (float)(baseVid + TRI_D[f][0][0]);
  const float a1 = (float)(baseVid + TRI_D[f][0][1]);
  const float a2 = (float)(baseVid + TRI_D[f][0][2]);
  const float b0 = (float)(baseVid + TRI_D[f][1][0]);
  const float b1 = (float)(baseVid + TRI_D[f][1][1]);
  const float b2 = (float)(baseVid + TRI_D[f][1][2]);

  const int j2 = tid * 2;                                   // first triangle

  // faces: 6 contiguous floats [a0 a1 a2 b0 b1 b2] at 6*tid (8B aligned)
  {
    float* fp = out + OFF_FACES + 6 * tid;
    st2(fp + 0, a0, a1);
    st2(fp + 2, a2, b0);
    st2(fp + 4, b1, b2);
  }
  // face_mask: repeat(exposed, 2)
  st2(out + OFF_FMASK + j2, m, m);

  // edge_index (2 x 6F), row-major. fwd = [(v0,v1)] ++ [(v1,v2)] ++ [(v0,v2)],
  // then reversed copies. Row 0 = sources at OFF_EDGE, row 1 at OFF_EDGE+6F.
  {
    float* e0 = out + OFF_EDGE;
    float* e1 = out + OFF_EDGE + 6 * FCNT;
    st2(e0 + 0 * FCNT + j2, a0, b0);  st2(e1 + 0 * FCNT + j2, a1, b1);
    st2(e0 + 1 * FCNT + j2, a1, b1);  st2(e1 + 1 * FCNT + j2, a2, b2);
    st2(e0 + 2 * FCNT + j2, a0, b0);  st2(e1 + 2 * FCNT + j2, a2, b2);
    st2(e0 + 3 * FCNT + j2, a1, b1);  st2(e1 + 3 * FCNT + j2, a0, b0);
    st2(e0 + 4 * FCNT + j2, a2, b2);  st2(e1 + 4 * FCNT + j2, a1, b1);
    st2(e0 + 5 * FCNT + j2, a2, b2);  st2(e1 + 5 * FCNT + j2, a0, b0);
  }
  // edge_mask = tile(face_mask, 6)
  {
    float* em = out + OFF_EMASK;
#pragma unroll
    for (int k = 0; k < 6; ++k) st2(em + k * FCNT + j2, m, m);
  }

  if (exposed) {
#pragma unroll
    for (int c = 0; c < 4; ++c)
      atomicOr(&wsUsed[baseVid + QUAD_D[f][c]], 1);
  }

  // exposed face contributes 2 triangles to face_counts[n]
  const float tot = wave_sum_wmma(exposed ? 2.0f : 0.0f);
  if ((threadIdx.x & 31) == 0)
    atomicAdd(&wsFaceCnt[n], (int)(tot + 0.5f));
}

// One thread per grid vertex; grid.y = batch so n is wave-uniform. Tail
// threads clamp (no early return) so EXEC is all-ones at the WMMA.
__global__ void k_verts(float* __restrict__ out,
                        const int* __restrict__ wsUsed,
                        int* __restrict__ wsVertCnt) {
  const int n  = blockIdx.y;
  const int lv = blockIdx.x * blockDim.x + threadIdx.x;
  const bool inb = lv < VG;
  int usedFlag = 0;
  if (inb) usedFlag = (wsUsed[n * VG + lv] != 0) ? 1 : 0;

  const float tot = wave_sum_wmma((float)usedFlag);
  if ((threadIdx.x & 31) == 0)
    atomicAdd(&wsVertCnt[n], (int)(tot + 0.5f));

  if (inb) {
    const int gv = n * VG + lv;
    const int gz = lv / 1089;
    const int r  = lv - gz * 1089;
    const int gy = r / 33;
    const int gx = r - gy * 33;
    out[OFF_USED + gv] = usedFlag ? 1.0f : 0.0f;
    float* vp = out + OFF_VPOS + 3 * gv;
    vp[0] = usedFlag ? ((float)gz - 0.5f) : 0.0f;
    vp[1] = usedFlag ? ((float)gy - 0.5f) : 0.0f;
    vp[2] = usedFlag ? ((float)gx - 0.5f) : 0.0f;
  }
}

__global__ void k_final(float* __restrict__ out,
                        const int* __restrict__ wsVert,
                        const int* __restrict__ wsFace) {
  const int i = threadIdx.x;
  if (i < N_) {
    out[OFF_VC + i] = (float)wsVert[i];
    out[OFF_FC + i] = (float)wsFace[i];
  }
}

extern "C" void kernel_launch(void* const* d_in, const int* in_sizes, int n_in,
                              void* d_out, int out_size, void* d_ws, size_t ws_size,
                              hipStream_t stream) {
  (void)in_sizes; (void)n_in; (void)out_size; (void)ws_size;
  const float* probas = (const float*)d_in[0];
  float* out = (float*)d_out;

  // workspace layout (ints): used bits [N*VG], vert counters [4], face counters [4]
  int* ws       = (int*)d_ws;
  int* wsUsed   = ws;
  int* wsVert   = ws + N_ * VG;
  int* wsFace   = ws + N_ * VG + N_;
  const int nInit = N_ * VG + 2 * N_;

  k_init <<<(nInit + 255) / 256, 256, 0, stream>>>(ws, nInit);
  k_faces<<<NPAIR / 256, 256, 0, stream>>>(probas, out, wsUsed, wsFace);
  k_verts<<<dim3((VG + 255) / 256, N_), 256, 0, stream>>>(out, wsUsed, wsVert);
  k_final<<<1, 64, 0, stream>>>(out, wsVert, wsFace);
}
